// VQCPCEncoder_3607772528761
// MI455X (gfx1250) — matrix operand order repaired and checked by reading, not compiled, not run on hardware
//
#include <hip/hip_runtime.h>
#include <hip/hip_bf16.h>
#include <math.h>

// Problem constants (match reference)
#define CH    768
#define ZD    64
#define CDIM  256
#define NCODE 512
#define BATCH 64
#define TP    1024                 // T' after stride-2 conv
#define NROWS (BATCH * TP)         // 65536 flattened (b, t') rows
#define KIM   320                  // im2col K = 80 * 4
#define GATES 1024                 // 4 * CDIM

typedef __attribute__((ext_vector_type(16))) __bf16 v16bf;
typedef __attribute__((ext_vector_type(8)))  float  v8f;
typedef __attribute__((ext_vector_type(4)))  unsigned int v4u;
typedef __attribute__((ext_vector_type(8)))  int v8i;
typedef __attribute__((ext_vector_type(4)))  int v4i;

struct __align__(16) U4 { unsigned int x, y, z, w; };
union Frag { struct { U4 a, b; } u; v16bf v; };

__device__ __forceinline__ unsigned short f32_to_bf16(float f) {
    unsigned int u = __float_as_uint(f);
    u += 0x7fffu + ((u >> 16) & 1u);          // round-to-nearest-even
    return (unsigned short)(u >> 16);
}
__device__ __forceinline__ float bf16_to_f32(unsigned short h) {
    return __uint_as_float(((unsigned int)h) << 16);
}
__device__ __forceinline__ float sigm(float x) { return 1.0f / (1.0f + __expf(-x)); }

// ---------------------------------------------------------------------------
// TDM: 2D tile load global -> LDS.
// Descriptor describes the tile itself as the tensor (tile==tensor, row
// stride = ld), so no OOB path is exercised. LDS pad: after every 16 DWORDs
// (32 bf16) insert 4 DWORDs (8 bf16) -> matches GK+8 row stride in LDS.
//   pad_interval code 3 -> 16 DWORDs ; pad_amount code 3 -> 4 DWORDs
// Toolchain here uses the 6-arg builtin:
//   (uint32x4 g0, int32x8 g1, int32x4 g2, int32x4 g3, int32x8 extra, i32 cpol)
// ---------------------------------------------------------------------------
__device__ __forceinline__ void tdm_load_2d_bf16(
    unsigned lds_byte_addr, const void* gaddr,
    unsigned tile_d0 /*elems*/, unsigned tile_d1 /*rows*/,
    unsigned long long ld /*row stride, elems*/)
{
    unsigned long long ga = (unsigned long long)gaddr;
    v4u g0;
    g0[0] = 1u;                                             // count = 1 (valid)
    g0[1] = lds_byte_addr;                                  // LDS dest
    g0[2] = (unsigned)(ga & 0xffffffffu);                   // global addr lo
    g0[3] = (unsigned)((ga >> 32) & 0x01ffffffu) | (2u << 30); // addr hi | type=2

    v8i g1;
    g1[0] = (int)((1u << 16)      // data_size = 2 bytes
                | (1u << 20)      // pad_enable
                | (3u << 22)      // pad_interval = 16 DWORDs
                | (3u << 25));    // pad_amount  = 4 DWORDs
    g1[1] = (int)((tile_d0 & 0xffffu) << 16);               // tensor_dim0 lo16
    g1[2] = (int)(((tile_d0 >> 16) & 0xffffu)               // tensor_dim0 hi16
                | ((tile_d1 & 0xffffu) << 16));             // tensor_dim1 lo16
    g1[3] = (int)(((tile_d1 >> 16) & 0xffffu)               // tensor_dim1 hi16
                | ((tile_d0 & 0xffffu) << 16));             // tile_dim0
    g1[4] = (int)(tile_d1 & 0xffffu);                       // tile_dim1 (dim2=0)
    g1[5] = (int)(unsigned)(ld & 0xffffffffu);              // dim0 stride lo32
    g1[6] = (int)((ld >> 32) & 0xffffu);                    // dim0 stride hi16
    g1[7] = 0;
    v4i z4 = {0, 0, 0, 0};
    v8i z8 = {0, 0, 0, 0, 0, 0, 0, 0};
    __builtin_amdgcn_tensor_load_to_lds(g0, g1, z4, z4, z8, 0);
}

// ---------------------------------------------------------------------------
// Elementwise fp32 -> bf16 (weight staging)
// ---------------------------------------------------------------------------
__global__ void cvt_bf16_kernel(const float* __restrict__ in,
                                unsigned short* __restrict__ out, int n) {
    int i = blockIdx.x * 256 + threadIdx.x;
    if (i < n) out[i] = f32_to_bf16(in[i]);
}

// bias = b_ih + b_hh ; also zero the loss accumulator
__global__ void bias_init_kernel(const float* __restrict__ b_ih,
                                 const float* __restrict__ b_hh,
                                 float* __restrict__ bias,
                                 float* __restrict__ loss) {
    int i = blockIdx.x * 256 + threadIdx.x;
    if (i < GATES) bias[i] = b_ih[i] + b_hh[i];
    if (i == 0) *loss = 0.0f;
}

// ---------------------------------------------------------------------------
// im2col for conv1d(k=4, s=2, p=1): out[n, cin*4+j] = mels[b, cin, 2t'-1+j]
// conv_w (OIH, flat ch*320 + cin*4 + j) already matches this K order.
// ---------------------------------------------------------------------------
__global__ void im2col_kernel(const float* __restrict__ mels,
                              unsigned short* __restrict__ out) {
    long i = (long)blockIdx.x * 256 + threadIdx.x;     // over NROWS*KIM
    int  kk = (int)(i % KIM);
    long n  = i / KIM;
    int  t  = (int)(n & (TP - 1));
    int  b  = (int)(n >> 10);
    int  cin = kk >> 2, j = kk & 3;
    int  pos = 2 * t - 1 + j;
    float v = (pos >= 0 && pos < 2048)
                  ? mels[((long)b * 80 + cin) * 2048 + pos] : 0.0f;
    out[i] = f32_to_bf16(v);
}

// ---------------------------------------------------------------------------
// Generic bf16 WMMA GEMM:  out[m, n] = sum_k A[m,k] * W[n,k] (+ bias[n])
//   A: [M, K] bf16 row-major   W: [Ncols, K] bf16 row-major (= B fragments)
// Block tile 128x64, 8 waves. Tiles staged by the Tensor Data Mover into
// double-buffered LDS (hardware pad to the +8 bf16 row stride); wave 0 issues
// the next tile's TDM before waiting on the current one (DMA/WMMA overlap).
// ---------------------------------------------------------------------------
#define GM 128
#define GN 64
#define GK 32
#define LDSTRIDE (GK + 8)

__global__ __launch_bounds__(256) void gemm_bf16_kernel(
    const unsigned short* __restrict__ A,
    const unsigned short* __restrict__ W,
    const float* __restrict__ bias,
    float* __restrict__ outF,
    unsigned short* __restrict__ outB,
    int K, int Ncols)
{
    __shared__ unsigned short sA[2][GM][LDSTRIDE];
    __shared__ unsigned short sW[2][GN][LDSTRIDE];

    const int  tid  = threadIdx.x;
    const int  lane = tid & 31;
    const int  wv   = tid >> 5;                // 0..7 -> 16-row strip
    const long rowBase = (long)blockIdx.y * GM;
    const int  colBase = blockIdx.x * GN;
    const bool issuer  = (wv == 0);            // wave 0 drives the TDM

    v8f acc[4] = {};

    // ISA fragment addressing (wave32)
    const int ar = lane & 15, ak = (lane >> 4) * 8;    // A: row, K-subgroup base
    const int bn = lane & 15, bk = (lane >> 4) * 8;    // B: col, K-subgroup base

    const unsigned ldsA[2] = { (unsigned)(unsigned long long)(void*)&sA[0][0][0],
                               (unsigned)(unsigned long long)(void*)&sA[1][0][0] };
    const unsigned ldsW[2] = { (unsigned)(unsigned long long)(void*)&sW[0][0][0],
                               (unsigned)(unsigned long long)(void*)&sW[1][0][0] };

    const int nsteps = K / GK;
    if (issuer) {   // prologue: stage first K-slice
        tdm_load_2d_bf16(ldsA[0], A + rowBase * (long)K, GK, GM, (unsigned long long)K);
        tdm_load_2d_bf16(ldsW[0], W + (long)colBase * K, GK, GN, (unsigned long long)K);
    }

    for (int s = 0; s < nsteps; ++s) {
        const int cur = s & 1, nxt = cur ^ 1;
        if (issuer) {
            if (s + 1 < nsteps) {   // stage next slice, then wait for current
                int ko = (s + 1) * GK;
                tdm_load_2d_bf16(ldsA[nxt], A + rowBase * (long)K + ko, GK, GM, (unsigned long long)K);
                tdm_load_2d_bf16(ldsW[nxt], W + (long)colBase * K + ko, GK, GN, (unsigned long long)K);
                __builtin_amdgcn_s_wait_tensorcnt(2);
            } else {
                __builtin_amdgcn_s_wait_tensorcnt(0);
            }
        }
        __syncthreads();    // current tiles visible to all waves

        const unsigned short (*tA)[LDSTRIDE] = sA[cur];
        const unsigned short (*tW)[LDSTRIDE] = sW[cur];
        Frag af;
        af.u.a = *(const U4*)&tA[wv * 16 + ar][ak];
        af.u.b = *(const U4*)&tA[wv * 16 + ar][ak + 16];
        Frag bf4[4];
#pragma unroll
        for (int ct = 0; ct < 4; ++ct) {
            bf4[ct].u.a = *(const U4*)&tW[ct * 16 + bn][bk];
            bf4[ct].u.b = *(const U4*)&tW[ct * 16 + bn][bk + 16];
        }
#pragma unroll
        for (int ct = 0; ct < 4; ++ct)
            acc[ct] = __builtin_amdgcn_wmma_f32_16x16x32_bf16(
                false, af.v, false, bf4[ct].v, (short)0, acc[ct], false, false);

        __syncthreads();    // reads done before this buffer is re-filled
    }

    // Epilogue: C layout -> lane 0-15: M=r; lane 16-31: M=r+8
#pragma unroll
    for (int ct = 0; ct < 4; ++ct) {
        int  n  = colBase + ct * 16 + (lane & 15);
        float bv = bias ? bias[n] : 0.0f;
#pragma unroll
        for (int e = 0; e < 8; ++e) {
            long m = rowBase + wv * 16 + e + (lane >> 4) * 8;
            float v = acc[ct][e] + bv;
            if (outF) outF[m * Ncols + n] = v;
            else      outB[m * Ncols + n] = f32_to_bf16(v);
        }
    }
}

// ---------------------------------------------------------------------------
// Fused LayerNorm(768) + ReLU, fp32 in -> bf16 out.  One wave per row.
// ---------------------------------------------------------------------------
__global__ __launch_bounds__(256) void ln_relu_kernel(
    const float* __restrict__ h, const float* __restrict__ g,
    const float* __restrict__ b, unsigned short* __restrict__ out)
{
    int row  = blockIdx.x * 8 + (threadIdx.x >> 5);
    int lane = threadIdx.x & 31;
    const float* x = h + (long)row * CH;
    float vals[24], s = 0.f, s2 = 0.f;
#pragma unroll
    for (int i = 0; i < 24; ++i) {
        float v = x[lane + i * 32];
        vals[i] = v; s += v; s2 += v * v;
    }
#pragma unroll
    for (int off = 16; off; off >>= 1) {
        s  += __shfl_xor(s,  off, 32);
        s2 += __shfl_xor(s2, off, 32);
    }
    float mean = s * (1.0f / CH);
    float var  = s2 * (1.0f / CH) - mean * mean;
    float inv  = rsqrtf(var + 1e-5f);
    unsigned short* o = out + (long)row * CH;
#pragma unroll
    for (int i = 0; i < 24; ++i) {
        int c = lane + i * 32;
        float v = (vals[i] - mean) * inv * g[c] + b[c];
        o[c] = f32_to_bf16(fmaxf(v, 0.0f));
    }
}

// ---------------------------------------------------------------------------
// VQ: argmin over 512 codes (dim 64). One wave per row; each lane scans 16
// codes; wave-reduce (value, index) with first-min tie-break like argmin.
// ---------------------------------------------------------------------------
__global__ __launch_bounds__(256) void vq_kernel(
    const float* __restrict__ h, const float* __restrict__ emb,
    float* __restrict__ qout, unsigned short* __restrict__ qbf,
    float* __restrict__ loss)
{
    __shared__ float sx[8][ZD];
    int wv   = threadIdx.x >> 5;
    int lane = threadIdx.x & 31;
    long row = (long)blockIdx.x * 8 + wv;

    const float* x = h + row * ZD;
    sx[wv][lane]      = x[lane];
    sx[wv][lane + 32] = x[lane + 32];
    __syncthreads();

    float best = 3.0e38f; int bidx = 0;
    for (int c = 0; c < NCODE / 32; ++c) {
        int m = c * 32 + lane;
        const float* e = emb + (long)m * ZD;
        float dot = 0.f, en = 0.f;
#pragma unroll 8
        for (int d = 0; d < ZD; ++d) { float ev = e[d]; dot += ev * sx[wv][d]; en += ev * ev; }
        float dist = en - 2.0f * dot;              // ||x||^2 constant per row
        if (dist < best) { best = dist; bidx = m; }
    }
#pragma unroll
    for (int off = 16; off; off >>= 1) {
        float ob = __shfl_xor(best, off, 32);
        int   oi = __shfl_xor(bidx, off, 32);
        if (ob < best || (ob == best && oi < bidx)) { best = ob; bidx = oi; }
    }

    const float* e = emb + (long)bidx * ZD;
    float psum = 0.f;
    for (int d = lane; d < ZD; d += 32) {
        float ev = e[d];
        qout[row * ZD + d] = ev;
        qbf [row * ZD + d] = f32_to_bf16(ev);
        float df = sx[wv][d] - ev;
        psum += df * df;
    }
#pragma unroll
    for (int off = 16; off; off >>= 1) psum += __shfl_xor(psum, off, 32);
    if (lane == 0)
        atomicAdd(loss, psum * (0.25f / ((float)NROWS * (float)ZD)));
}

// ---------------------------------------------------------------------------
// Persistent single-workgroup LSTM scan (1024 steps, B=64, CD=256).
// 32 waves; wave w owns hidden-col tile ct = w>>1 (16 cds) for ALL FOUR gates
// and row-half rh = w&1 -> i/f/g/o combine entirely in-wave in C-frag layout.
// h state: LDS bf16. c state: registers. w_hh streams from L2 (bf16).
// ---------------------------------------------------------------------------
__global__ __launch_bounds__(1024) void lstm_kernel(
    const unsigned short* __restrict__ xs,    // [NROWS, 1024] bf16
    const unsigned short* __restrict__ whh,   // [1024, 256] bf16
    float* __restrict__ cseq)                 // [B, TP, CD] fp32
{
    __shared__ unsigned short hprev[BATCH][CDIM + 8];

    const int tid  = threadIdx.x;
    const int lane = tid & 31;
    const int wv   = tid >> 5;
    const int ct   = wv >> 1;          // 0..15 hidden-col tile
    const int rh   = wv & 1;           // row half (rows rh*32 .. rh*32+31)

    for (int i = tid; i < BATCH * (CDIM + 8); i += 1024)
        ((unsigned short*)hprev)[i] = 0;

    v8f cst[2] = {};                   // c state fragments (2 row tiles)

    const int an = lane & 15, ak = (lane >> 4) * 8;
    const int bn = lane & 15, bk = (lane >> 4) * 8;
    __syncthreads();

    for (int t = 0; t < TP; ++t) {
        v8f acc[4][2] = {};            // [gate][row tile]
        for (int ko = 0; ko < CDIM; ko += 32) {
            Frag af[2];
#pragma unroll
            for (int rt = 0; rt < 2; ++rt) {
                int r = rh * 32 + rt * 16 + an;
                af[rt].u.a = *(const U4*)&hprev[r][ko + ak];
                af[rt].u.b = *(const U4*)&hprev[r][ko + ak + 16];
            }
#pragma unroll
            for (int g = 0; g < 4; ++g) {
                int col = g * CDIM + ct * 16 + bn;
                const U4* wp = (const U4*)(whh + (long)col * CDIM + ko + bk);
                Frag bf_;
                bf_.u.a = wp[0];
                bf_.u.b = wp[2];
#pragma unroll
                for (int rt = 0; rt < 2; ++rt)
                    acc[g][rt] = __builtin_amdgcn_wmma_f32_16x16x32_bf16(
                        false, af[rt].v, false, bf_.v, (short)0, acc[g][rt], false, false);
            }
        }
        __syncthreads();               // everyone done reading hprev

        if (t + 1 < TP) {              // near-cache prefetch of next step's xs
            long pb = ((long)(rh * 32 + (lane & 15)) * TP + t + 1) * GATES + ct * 16;
            __builtin_prefetch((const void*)&xs[pb], 0, 3);
        }

#pragma unroll
        for (int rt = 0; rt < 2; ++rt) {
#pragma unroll
            for (int e = 0; e < 8; ++e) {
                int  rowm  = rh * 32 + rt * 16 + e + (lane >> 4) * 8;   // batch idx
                int  cd    = ct * 16 + (lane & 15);
                long xbase = ((long)rowm * TP + t) * GATES;
                float iv = acc[0][rt][e] + bf16_to_f32(xs[xbase + 0 * CDIM + cd]);
                float fv = acc[1][rt][e] + bf16_to_f32(xs[xbase + 1 * CDIM + cd]);
                float gv = acc[2][rt][e] + bf16_to_f32(xs[xbase + 2 * CDIM + cd]);
                float ov = acc[3][rt][e] + bf16_to_f32(xs[xbase + 3 * CDIM + cd]);
                float cn = sigm(fv) * cst[rt][e] + sigm(iv) * tanhf(gv);
                float hn = sigm(ov) * tanhf(cn);
                cst[rt][e] = cn;
                hprev[rowm][cd] = f32_to_bf16(hn);
                cseq[((long)rowm * TP + t) * CDIM + cd] = hn;
            }
        }
        __syncthreads();               // h_n visible before next step's reads
    }
}

// ---------------------------------------------------------------------------
// Host launcher
// ---------------------------------------------------------------------------
extern "C" void kernel_launch(void* const* d_in, const int* in_sizes, int n_in,
                              void* d_out, int out_size, void* d_ws, size_t ws_size,
                              hipStream_t stream)
{
    (void)in_sizes; (void)n_in; (void)out_size; (void)ws_size;

    const float* mels   = (const float*)d_in[0];
    const float* conv_w = (const float*)d_in[1];
    const float* ln_g   = (const float*)d_in[2];
    const float* ln_b   = (const float*)d_in[3];
    const float* lin_w  = (const float*)d_in[4];
    const float* w_out  = (const float*)d_in[5];
    const float* emb    = (const float*)d_in[6];
    const float* w_ih   = (const float*)d_in[7];
    const float* w_hh   = (const float*)d_in[8];
    const float* b_ih   = (const float*)d_in[9];
    const float* b_hh   = (const float*)d_in[10];

    float* q_out = (float*)d_out;                       // [N, 64]
    float* c_seq = q_out + (size_t)NROWS * ZD;          // [B, TP, 256]
    float* loss  = c_seq + (size_t)NROWS * CDIM;        // [1]

    char* p = (char*)d_ws;
    auto alloc = [&](size_t bytes) {
        char* r = p; p += (bytes + 255) & ~(size_t)255; return r;
    };
    unsigned short* act  = (unsigned short*)alloc((size_t)NROWS * CH * 2);   // also im2col
    float*          hbuf = (float*)         alloc((size_t)NROWS * CH * 4);
    float*          hsml = (float*)         alloc((size_t)NROWS * ZD * 4);
    unsigned short* qbf  = (unsigned short*)alloc((size_t)NROWS * ZD * 2);
    unsigned short* xsb  = (unsigned short*)alloc((size_t)NROWS * GATES * 2);
    unsigned short* wc   = (unsigned short*)alloc((size_t)CH * KIM * 2);
    unsigned short* wlin = (unsigned short*)alloc((size_t)4 * CH * CH * 2);
    unsigned short* wob  = (unsigned short*)alloc((size_t)ZD * CH * 2);
    unsigned short* wihb = (unsigned short*)alloc((size_t)GATES * ZD * 2);
    unsigned short* whhb = (unsigned short*)alloc((size_t)GATES * CDIM * 2);
    float*          bias = (float*)         alloc(GATES * 4);

    auto cvt = [&](const float* src, unsigned short* dst, int n) {
        cvt_bf16_kernel<<<dim3((n + 255) / 256), dim3(256), 0, stream>>>(src, dst, n);
    };
    cvt(conv_w, wc,   CH * KIM);
    cvt(lin_w,  wlin, 4 * CH * CH);
    cvt(w_out,  wob,  ZD * CH);
    cvt(w_ih,   wihb, GATES * ZD);
    cvt(w_hh,   whhb, GATES * CDIM);
    bias_init_kernel<<<dim3(4), dim3(256), 0, stream>>>(b_ih, b_hh, bias, loss);

    // conv as im2col + GEMM
    im2col_kernel<<<dim3((NROWS * KIM) / 256), dim3(256), 0, stream>>>(mels, act);
    gemm_bf16_kernel<<<dim3(CH / GN, NROWS / GM), dim3(256), 0, stream>>>(
        act, wc, nullptr, hbuf, nullptr, KIM, CH);

    // 4x (LN -> ReLU -> 768x768)
    for (int i = 0; i < 4; ++i) {
        ln_relu_kernel<<<dim3(NROWS / 8), dim3(256), 0, stream>>>(
            hbuf, ln_g + i * CH, ln_b + i * CH, act);
        gemm_bf16_kernel<<<dim3(CH / GN, NROWS / GM), dim3(256), 0, stream>>>(
            act, wlin + (size_t)i * CH * CH, nullptr, hbuf, nullptr, CH, CH);
    }

    // final LN -> ReLU -> 768x64
    ln_relu_kernel<<<dim3(NROWS / 8), dim3(256), 0, stream>>>(
        hbuf, ln_g + 4 * CH, ln_b + 4 * CH, act);
    gemm_bf16_kernel<<<dim3(ZD / GN, NROWS / GM), dim3(256), 0, stream>>>(
        act, wob, nullptr, hsml, nullptr, CH, ZD);

    // VQ
    vq_kernel<<<dim3(NROWS / 8), dim3(256), 0, stream>>>(hsml, emb, q_out, qbf, loss);

    // LSTM input projection: xs = q @ w_ih^T + (b_ih + b_hh), bf16 out
    gemm_bf16_kernel<<<dim3(GATES / GN, NROWS / GM), dim3(256), 0, stream>>>(
        qbf, wihb, bias, nullptr, xsb, ZD, GATES);

    // Recurrent scan (sequential): single persistent workgroup
    lstm_kernel<<<dim3(1), dim3(1024), 0, stream>>>(xsb, whhb, c_seq);
}
